// Phonology_25460566130712
// MI455X (gfx1250) — compile-verified, matrix operand order checked by reading, not run on hardware
//
#include <hip/hip_runtime.h>

#define B_ 2048
#define F_ 64
#define R_ 32
#define P_ 16
#define C_ 128
#define TAU_ 2.0f

typedef __attribute__((ext_vector_type(16))) _Float16 v16h;
typedef __attribute__((ext_vector_type(8)))  _Float16 v8h;
typedef __attribute__((ext_vector_type(8)))  float    v8f;

// ---------------------------------------------------------------- utilities
__device__ __forceinline__ float sigmoidf_(float x) {
  return 1.0f / (1.0f + __expf(-x));
}

// softmax_r( -TAU*(posn - r)^2 ), evaluated for one r, max-shifted for stability
__device__ __forceinline__ float attn_w(float posn, float rf) {
  float rn = fminf(fmaxf(roundf(posn), 0.0f), 31.0f);
  float dm = posn - rn;
  float m  = -TAU_ * dm * dm;
  float denom = 0.0f;
#pragma unroll
  for (int j = 0; j < R_; ++j) {
    float d = posn - (float)j;
    denom += __expf(-TAU_ * d * d - m);
  }
  float d = posn - rf;
  return __expf(-TAU_ * d * d - m) / denom;
}

// ---------------------------------------------------------------- f32 -> f16
__global__ void cvt_f32_f16(const float* __restrict__ src,
                            _Float16* __restrict__ dst, int n) {
  int i = blockIdx.x * blockDim.x + threadIdx.x;
  if (i < n) dst[i] = (_Float16)src[i];
}

// --------------------------------------- f32 (C x N) -> f16 transposed (N x C)
// Makes WMMA B-fragment loads contiguous (2 x b128 per k-step like A).
__global__ void cvt_t_f16(const float* __restrict__ src,
                          _Float16* __restrict__ dst, int N) {
  int i = blockIdx.x * blockDim.x + threadIdx.x;  // over C_*N, src-coalesced
  if (i >= C_ * N) return;
  int c = i / N;
  int n = i - c * N;
  dst[(size_t)n * C_ + c] = (_Float16)src[i];
}

// ---------------------------------------------------- WMMA GEMM: Y = A@W + b
// A : (B_, 128) f16 row-major;  Wt : (N, 128) f16 (W transposed);
// Y : (B_, N) f32.  One wave per 16x16 tile; K=128 -> 4 WMMA.
__global__ void wmma_lin(const _Float16* __restrict__ Ah,
                         const _Float16* __restrict__ Wt,
                         const float* __restrict__ bias,
                         float* __restrict__ Y, int N) {
  const int lane = threadIdx.x;          // 0..31 (wave32)
  const int row0 = blockIdx.x * 16;      // M tile
  const int col0 = blockIdx.y * 16;      // N tile
  const int lr = lane & 15;
  const int kh = (lane >> 4) * 8;        // 16-bit A/B lane-half K offset
  const _Float16* arow = Ah + (size_t)(row0 + lr) * C_ + kh;
  const _Float16* brow = Wt + (size_t)(col0 + lr) * C_ + kh;
  v8f acc = {};
#pragma unroll
  for (int kk = 0; kk < 4; ++kk) {
    const int k0 = kk * 32;
    if (kk + 1 < 4) {  // speculative prefetch of next K block (safe: dropped on fault)
      __builtin_prefetch(arow + k0 + 32, 0, 0);
      __builtin_prefetch(brow + k0 + 32, 0, 0);
    }
    // Fragment layout (ISA 16-bit 16x32): lane-half kh, halves j -> K = k0+kh+j
    // for j<8 and K = k0+16+kh+(j-8) for j>=8. Two contiguous 16B chunks each.
    v8h alo = *(const v8h*)(arow + k0);
    v8h ahi = *(const v8h*)(arow + k0 + 16);
    v8h blo = *(const v8h*)(brow + k0);
    v8h bhi = *(const v8h*)(brow + k0 + 16);
    v16h a, b;
#pragma unroll
    for (int j = 0; j < 8; ++j) {
      a[j] = alo[j]; a[8 + j] = ahi[j];
      b[j] = blo[j]; b[8 + j] = bhi[j];
    }
    acc = __builtin_amdgcn_wmma_f32_16x16x32_f16(
        false, a, false, b, (short)0, acc, false, false);
  }
  // C/D layout: lane = n, VGPR v -> M = row0 + v + 8*(lane>=16)
  const int n = col0 + lr;
  const float bn = bias[n];
  const int mbase = row0 + (lane >> 4) * 8;
#pragma unroll
  for (int v = 0; v < 8; ++v)
    Y[(size_t)(mbase + v) * N + n] = acc[v] + bn;
}

// ------------------------------------------------------- N==1 linears (dot)
__global__ void dot_c(const float* __restrict__ ctx,
                      const float* __restrict__ W,
                      const float* __restrict__ b0,
                      float* __restrict__ y) {
  int b = blockIdx.x * blockDim.x + threadIdx.x;
  if (b >= B_) return;
  float s = b0[0];
#pragma unroll 8
  for (int c = 0; c < C_; ++c) s += ctx[(size_t)b * C_ + c] * W[c];
  y[b] = s;
}

// ------------------------------------------------------------- match kernel
// match[b,p,r] = exp( sum_k log_sigmoid( sum_f Wk[k,b,p,f]*form[b,f,r+k-1] + bk ) )
__global__ void match_kernel(const float* __restrict__ form,  // B,F,R
                             const float* __restrict__ Ym,    // 3,B,1024
                             const float* __restrict__ Ymb,   // 3,B,16
                             float* __restrict__ match) {     // B,P,R
  const int b = blockIdx.x;
  const int t = threadIdx.x;  // 512 threads: (p,r)
  __shared__ float sform[F_ * R_];
  __shared__ float sW[3 * P_ * F_];
  for (int i = t; i < F_ * R_; i += 512)
    sform[i] = form[(size_t)b * F_ * R_ + i];
  for (int i = t; i < 3 * P_ * F_; i += 512)
    sW[i] = Ym[(size_t)(i >> 10) * B_ * 1024 + (size_t)b * 1024 + (i & 1023)];
  __syncthreads();
  const int p = t >> 5;
  const int r = t & 31;
  float acc = 0.0f;
#pragma unroll
  for (int k = 0; k < 3; ++k) {
    int rs = r + k - 1;  // prev / cur / next with zero padding
    float s = Ymb[(size_t)k * B_ * P_ + (size_t)b * P_ + p];
    if (rs >= 0 && rs < R_) {
      const float* w  = &sW[k * P_ * F_ + p * F_];
      const float* fr = &sform[rs];
#pragma unroll 8
      for (int f = 0; f < F_; ++f) s += w[f] * fr[f * R_];
    }
    // stable log_sigmoid(s) = min(s,0) - log1p(exp(-|s|))
    acc += fminf(s, 0.0f) - log1pf(__expf(-fabsf(s)));
  }
  match[(size_t)b * P_ * R_ + p * R_ + r] = __expf(acc);
}

// --------------------------------------------------------------- scan kernel
// One block per batch; out/form/weights in LDS; serial over R=32 steps.
__global__ void scan_kernel(const float* __restrict__ form,   // B,F,R
                            const float* __restrict__ match,  // B,P,R
                            const float* __restrict__ Ymod,   // B,96
                            const float* __restrict__ Ywch,   // B,1024
                            const float* __restrict__ Ywdl,   // B,16
                            const float* __restrict__ Ywep,   // B,64
                            const float* __restrict__ Yvch,   // B,1024
                            const float* __restrict__ Yvep,   // B,4096
                            const float* __restrict__ Ywnc,   // B,64
                            const float* __restrict__ Ywnd,   // B
                            const float* __restrict__ Ywne,   // B
                            float* __restrict__ out) {        // B,F,R
  const int b = blockIdx.x;
  const int t = threadIdx.x;  // 256 threads
  __shared__ float s_mod[P_ * 6];
  __shared__ float s_match[P_ * R_];
  __shared__ float s_wch[P_ * F_];
  __shared__ float s_vch[P_ * F_];
  __shared__ float s_vep[P_ * F_ * 4];
  __shared__ float s_wep[P_ * 4];
  __shared__ float s_wdl[P_];
  __shared__ float s_wnc[F_];
  __shared__ float s_form[F_ * R_];
  __shared__ float s_out[F_ * R_];
  __shared__ float s_v[F_];
  __shared__ float s_sym[F_];
  __shared__ float s_sc[2];  // exp(wnd), exp(wne)

  if (t < P_) {  // softmax over the 6 modulation logits
    float l[6], mx = -1e30f;
#pragma unroll
    for (int j = 0; j < 6; ++j) {
      l[j] = Ymod[(size_t)b * 96 + t * 6 + j];
      mx = fmaxf(mx, l[j]);
    }
    float s = 0.0f;
#pragma unroll
    for (int j = 0; j < 6; ++j) { l[j] = __expf(l[j] - mx); s += l[j]; }
#pragma unroll
    for (int j = 0; j < 6; ++j) s_mod[t * 6 + j] = l[j] / s;
  }
  if (t == 0) {
    s_sc[0] = __expf(Ywnd[b]);
    s_sc[1] = __expf(Ywne[b]);
  }
  __syncthreads();
  for (int i = t; i < P_ * F_; i += 256) {
    s_wch[i] = s_mod[(i >> 6) * 6 + 0] * Ywch[(size_t)b * 1024 + i];
    s_vch[i] = Yvch[(size_t)b * 1024 + i];
  }
  for (int i = t; i < P_ * F_ * 4; i += 256)
    s_vep[i] = Yvep[(size_t)b * 4096 + i];
  if (t < P_ * 4)
    s_wep[t] = s_mod[(t >> 2) * 6 + 2 + (t & 3)] * Ywep[(size_t)b * 64 + t];
  if (t < P_) s_wdl[t] = s_mod[t * 6 + 1] * Ywdl[(size_t)b * 16 + t];
  if (t < F_) s_wnc[t] = __expf(Ywnc[(size_t)b * 64 + t]);
  for (int i = t; i < P_ * R_; i += 256)
    s_match[i] = match[(size_t)b * P_ * R_ + i];
  for (int i = t; i < F_ * R_; i += 256) {
    s_form[i] = form[(size_t)b * F_ * R_ + i];
    s_out[i] = 0.0f;
  }
  __syncthreads();

  const int r = t & 31;
  const int fg = t >> 5;       // 8 feature groups of 8
  const float rf = (float)r;
  float posn = 0.0f;           // identical in every thread (redundant scalars)
  const float wnd = s_sc[0];
  const float wne = s_sc[1];

  for (int s = 0; s < R_; ++s) {
    // ---- epen, endpoint index 0
    float S0 = 0.0f;
    for (int p = 0; p < P_; ++p) S0 += s_match[p * R_ + s] * s_wep[p * 4 + 0];
    float p0 = sigmoidf_(S0 - wne);
    if (t < F_) {
      float v = 0.0f;
      for (int p = 0; p < P_; ++p) {
        float we = s_match[p * R_ + s] * s_wep[p * 4 + 0];
        v += fmaxf(we, 0.0f) * s_vep[p * 256 + t * 4 + 0];
      }
      s_v[t] = v;
    }
    __syncthreads();
    {
      float a = attn_w(posn, rf) * p0;
#pragma unroll
      for (int i = 0; i < 8; ++i) {
        int idx = (fg * 8 + i) * R_ + r;
        float o = s_out[idx];
        s_out[idx] = o + a * (s_v[fg * 8 + i] - o);  // fused bind+mix
      }
    }
    posn += p0;

    // ---- chng (per-f) + deln (scalar)
    if (t < F_) {
      float sumw = 0.0f, vc = 0.0f;
      for (int p = 0; p < P_; ++p) {
        float wc = s_match[p * R_ + s] * s_wch[p * F_ + t];
        sumw += wc;
        vc += fmaxf(wc, 0.0f) * s_vch[p * F_ + t];
      }
      float pch = sigmoidf_(sumw - s_wnc[t]);
      float si = s_form[t * R_ + s];
      s_sym[t] = si + pch * (vc - si);
    }
    float Sd = 0.0f;
    for (int p = 0; p < P_; ++p) Sd += s_match[p * R_ + s] * s_wdl[p];
    float pd = sigmoidf_(Sd - wnd);
    __syncthreads();
    {
      float a = attn_w(posn, rf) * (1.0f - pd);
#pragma unroll
      for (int i = 0; i < 8; ++i) {
        int idx = (fg * 8 + i) * R_ + r;
        float o = s_out[idx];
        s_out[idx] = o + a * (s_sym[fg * 8 + i] - o);
      }
    }
    posn += (1.0f - pd);
    __syncthreads();

    // ---- epen, endpoint index 2
    float S2 = 0.0f;
    for (int p = 0; p < P_; ++p) S2 += s_match[p * R_ + s] * s_wep[p * 4 + 2];
    float p2 = sigmoidf_(S2 - wne);
    if (t < F_) {
      float v = 0.0f;
      for (int p = 0; p < P_; ++p) {
        float we = s_match[p * R_ + s] * s_wep[p * 4 + 2];
        v += fmaxf(we, 0.0f) * s_vep[p * 256 + t * 4 + 2];
      }
      s_v[t] = v;
    }
    __syncthreads();
    {
      float a = attn_w(posn, rf) * p2;
#pragma unroll
      for (int i = 0; i < 8; ++i) {
        int idx = (fg * 8 + i) * R_ + r;
        float o = s_out[idx];
        s_out[idx] = o + a * (s_v[fg * 8 + i] - o);
      }
    }
    posn += p2;
    __syncthreads();
  }
  for (int i = t; i < F_ * R_; i += 256)
    out[(size_t)b * F_ * R_ + i] = s_out[i];
}

// ------------------------------------------------------------------- launch
extern "C" void kernel_launch(void* const* d_in, const int* in_sizes, int n_in,
                              void* d_out, int out_size, void* d_ws,
                              size_t ws_size, hipStream_t stream) {
  (void)in_sizes; (void)n_in; (void)out_size; (void)ws_size;
  const float* form  = (const float*)d_in[0];
  const float* ctx   = (const float*)d_in[1];
  const float* mod_W = (const float*)d_in[2];
  const float* mod_b = (const float*)d_in[3];
  const float* wch_W = (const float*)d_in[4];
  const float* wch_b = (const float*)d_in[5];
  const float* wdl_W = (const float*)d_in[6];
  const float* wdl_b = (const float*)d_in[7];
  const float* wep_W = (const float*)d_in[8];
  const float* wep_b = (const float*)d_in[9];
  const float* vch_W = (const float*)d_in[10];
  const float* vch_b = (const float*)d_in[11];
  const float* vep_W = (const float*)d_in[12];
  const float* vep_b = (const float*)d_in[13];
  const float* wnc_W = (const float*)d_in[14];
  const float* wnc_b = (const float*)d_in[15];
  const float* wnd_W = (const float*)d_in[16];
  const float* wnd_b = (const float*)d_in[17];
  const float* wne_W = (const float*)d_in[18];
  const float* wne_b = (const float*)d_in[19];
  const float* m_W   = (const float*)d_in[20];
  const float* m_Wb  = (const float*)d_in[21];
  const float* mb_W  = (const float*)d_in[22];
  const float* mb_b  = (const float*)d_in[23];
  float* out = (float*)d_out;

  char* ws = (char*)d_ws;
  size_t off = 0;
  auto take = [&](size_t bytes) -> void* {
    void* p = (void*)(ws + off);
    off = (off + bytes + 255) & ~(size_t)255;
    return p;
  };
  // f16 staging (weights transposed to N x C)
  _Float16* ctxh  = (_Float16*)take((size_t)B_ * C_ * 2);
  _Float16* h_mod = (_Float16*)take((size_t)C_ * 96 * 2);
  _Float16* h_wch = (_Float16*)take((size_t)C_ * 1024 * 2);
  _Float16* h_wdl = (_Float16*)take((size_t)C_ * 16 * 2);
  _Float16* h_wep = (_Float16*)take((size_t)C_ * 64 * 2);
  _Float16* h_vch = (_Float16*)take((size_t)C_ * 1024 * 2);
  _Float16* h_vep = (_Float16*)take((size_t)C_ * 4096 * 2);
  _Float16* h_wnc = (_Float16*)take((size_t)C_ * 64 * 2);
  _Float16* h_m   = (_Float16*)take((size_t)3 * C_ * 1024 * 2);
  _Float16* h_mb  = (_Float16*)take((size_t)3 * C_ * 16 * 2);
  // f32 GEMM outputs
  float* Y_mod = (float*)take((size_t)B_ * 96 * 4);
  float* Y_wch = (float*)take((size_t)B_ * 1024 * 4);
  float* Y_wdl = (float*)take((size_t)B_ * 16 * 4);
  float* Y_wep = (float*)take((size_t)B_ * 64 * 4);
  float* Y_vch = (float*)take((size_t)B_ * 1024 * 4);
  float* Y_vep = (float*)take((size_t)B_ * 4096 * 4);
  float* Y_wnc = (float*)take((size_t)B_ * 64 * 4);
  float* Y_wnd = (float*)take((size_t)B_ * 4);
  float* Y_wne = (float*)take((size_t)B_ * 4);
  float* Y_m   = (float*)take((size_t)3 * B_ * 1024 * 4);
  float* Y_mb  = (float*)take((size_t)3 * B_ * 16 * 4);
  float* matc  = (float*)take((size_t)B_ * P_ * R_ * 4);

  // context: plain convert (row-major B x C is already what WMMA A wants)
  cvt_f32_f16<<<(B_ * C_ + 255) / 256, 256, 0, stream>>>(ctx, ctxh, B_ * C_);
  // weights: convert + transpose (C x N -> N x C)
  auto cvt = [&](const float* s, _Float16* d, int N) {
    cvt_t_f16<<<(C_ * N + 255) / 256, 256, 0, stream>>>(s, d, N);
  };
  cvt(mod_W, h_mod, 96);
  cvt(wch_W, h_wch, 1024);
  cvt(wdl_W, h_wdl, 16);
  cvt(wep_W, h_wep, 64);
  cvt(vch_W, h_vch, 1024);
  cvt(vep_W, h_vep, 4096);
  cvt(wnc_W, h_wnc, 64);
  for (int k = 0; k < 3; ++k) {
    cvt(m_W + (size_t)k * C_ * 1024, h_m + (size_t)k * 1024 * C_, 1024);
    cvt(mb_W + (size_t)k * C_ * 16, h_mb + (size_t)k * 16 * C_, 16);
  }

  auto gemm = [&](const _Float16* Wt, const float* bias, float* Y, int N) {
    wmma_lin<<<dim3(B_ / 16, N / 16), 32, 0, stream>>>(ctxh, Wt, bias, Y, N);
  };
  gemm(h_mod, mod_b, Y_mod, 96);
  gemm(h_wch, wch_b, Y_wch, 1024);
  gemm(h_wdl, wdl_b, Y_wdl, 16);
  gemm(h_wep, wep_b, Y_wep, 64);
  gemm(h_vch, vch_b, Y_vch, 1024);
  gemm(h_vep, vep_b, Y_vep, 4096);
  gemm(h_wnc, wnc_b, Y_wnc, 64);
  for (int k = 0; k < 3; ++k) {
    gemm(h_m + (size_t)k * 1024 * C_, m_Wb + (size_t)k * 1024,
         Y_m + (size_t)k * B_ * 1024, 1024);
    gemm(h_mb + (size_t)k * 16 * C_, mb_b + (size_t)k * 16,
         Y_mb + (size_t)k * B_ * 16, 16);
  }
  dot_c<<<(B_ + 255) / 256, 256, 0, stream>>>(ctx, wnd_W, wnd_b, Y_wnd);
  dot_c<<<(B_ + 255) / 256, 256, 0, stream>>>(ctx, wne_W, wne_b, Y_wne);

  match_kernel<<<B_, 512, 0, stream>>>(form, Y_m, Y_mb, matc);
  scan_kernel<<<B_, 256, 0, stream>>>(form, matc, Y_mod, Y_wch, Y_wdl, Y_wep,
                                      Y_vch, Y_vep, Y_wnc, Y_wnd, Y_wne, out);
}
